// Decoder_70952859730160
// MI455X (gfx1250) — matrix operand order repaired and checked by reading, not compiled, not run on hardware
//
#include <hip/hip_runtime.h>
#include <hip/hip_bf16.h>

// ============================================================================
// Decoder with additive attention + GRUCell, S=384 sequential steps.
//
// Optimization insight: energy = (h@u_h + c0) + score_e, with the h-term a
// per-batch scalar broadcast over the sequence axis. Softmax over the sequence
// axis is shift-invariant, so alpha (and therefore ctx) is CONSTANT across all
// steps. We precompute alpha/ctx once and fold all ctx-dependent linear terms
// into per-step bias tables (gctx, octx). The recurrent loop reduces to small
// bf16 WMMA GEMMs + GRU gate math + a 256-wide softmax, executed by a single
// persistent kernel (16 WGs x 256 thr) with LDS-resident weight tiles and a
// global sense-reversing barrier (3 syncs/step).
// ============================================================================

#define S_LEN 384
#define B_N   128
#define H_N   256
#define NWG   16
#define NTHR  256

typedef float  v8f   __attribute__((ext_vector_type(8)));
typedef __bf16 v16bf __attribute__((ext_vector_type(16)));

union BFrag { uint4 q[2]; v16bf v; };

__device__ __forceinline__ unsigned short f2bf(float f) {
  union { float f; unsigned u; } x; x.f = f;
  unsigned r = x.u + 0x7FFFu + ((x.u >> 16) & 1u);   // round-to-nearest-even
  return (unsigned short)(r >> 16);
}

// A fragment (16x32, bf16): X row-major [*, ld], rows m0..m0+15, K = k0..k0+31.
// ISA layout: lane (m=l&15, g=l>>4): elems e0..7 -> K=8g+e, e8..15 -> K=16+8g+(e-8).
__device__ __forceinline__ v16bf load_frag_a(const unsigned short* X, int ld,
                                             int m0, int k0, int lane) {
  int m = lane & 15, g = lane >> 4;
  const unsigned short* p = X + (size_t)(m0 + m) * ld + k0 + g * 8;
  BFrag f;
  f.q[0] = *(const uint4*)(p);        // K = 8g .. 8g+7
  f.q[1] = *(const uint4*)(p + 16);   // K = 16+8g .. 16+8g+7
  return f.v;
}

// B fragment (32x16, bf16): B[k,n] = W[n0+n, k], W row-major [N, ld].
// ISA layout: lane (n=l&15, g=l>>4): elems e0..7 -> K=8g+e, e8..15 -> K=16+8g+(e-8).
__device__ __forceinline__ v16bf load_frag_b(const unsigned short* W, int ld,
                                             int n0, int k0, int lane) {
  int n = lane & 15, g = lane >> 4;
  const unsigned short* p = W + (size_t)(n0 + n) * ld + k0 + g * 8;
  BFrag f;
  f.q[0] = *(const uint4*)(p);
  f.q[1] = *(const uint4*)(p + 16);
  return f.v;
}

__device__ __forceinline__ v8f wmma_bf16(v16bf a, v16bf b, v8f c) {
  return __builtin_amdgcn_wmma_f32_16x16x32_bf16(false, a, false, b,
                                                 (short)0, c, false, false);
}

__device__ __forceinline__ float wave_sum(float v) {
  #pragma unroll
  for (int m = 16; m >= 1; m >>= 1) v += __shfl_xor(v, m, 32);
  return v;
}
__device__ __forceinline__ float wave_max(float v) {
  #pragma unroll
  for (int m = 16; m >= 1; m >>= 1) v = fmaxf(v, __shfl_xor(v, m, 32));
  return v;
}

__device__ __forceinline__ float sigmoidf_(float x) { return 1.0f / (1.0f + __expf(-x)); }
__device__ __forceinline__ float tanhf_(float x)    { return 1.0f - 2.0f / (__expf(2.0f * x) + 1.0f); }

// Grid-wide sense-reversing barrier, agent scope. State zeroed per launch.
__device__ __forceinline__ void grid_barrier(unsigned* cnt, unsigned* gen) {
  __syncthreads();
  if (threadIdx.x == 0) {
    unsigned g = __hip_atomic_load(gen, __ATOMIC_ACQUIRE, __HIP_MEMORY_SCOPE_AGENT);
    unsigned a = __hip_atomic_fetch_add(cnt, 1u, __ATOMIC_ACQ_REL, __HIP_MEMORY_SCOPE_AGENT);
    if (a == NWG - 1) {
      __hip_atomic_store(cnt, 0u, __ATOMIC_RELAXED, __HIP_MEMORY_SCOPE_AGENT);
      __hip_atomic_fetch_add(gen, 1u, __ATOMIC_ACQ_REL, __HIP_MEMORY_SCOPE_AGENT);
    } else {
      while (__hip_atomic_load(gen, __ATOMIC_ACQUIRE, __HIP_MEMORY_SCOPE_AGENT) == g)
        __builtin_amdgcn_s_sleep(1);
    }
  }
  __syncthreads();
}

// ---------------------------------------------------------------------------
// Precompute kernels (one-shot, memory bound)
// ---------------------------------------------------------------------------

// u_e[j] = sum_i Wa[i, 256+j] * v[i]
__global__ void k_ue(const float* __restrict__ Wa, const float* __restrict__ v,
                     float* __restrict__ u_e) {
  int j = threadIdx.x;
  float acc = 0.f;
  for (int i = 0; i < 256; ++i) acc += Wa[i * 512 + 256 + j] * v[i];
  u_e[j] = acc;
}

// score[s*B+b] = E[s,b,:] . u_e  (one wave per row)
__global__ void k_score(const float* __restrict__ E, const float* __restrict__ u_e,
                        float* __restrict__ score) {
  int w = (blockIdx.x << 3) + (threadIdx.x >> 5);
  int lane = threadIdx.x & 31;
  const float* row = E + (size_t)w * 256;
  float acc = 0.f;
  for (int h = lane; h < 256; h += 32) acc += row[h] * u_e[h];
  acc = wave_sum(acc);
  if (lane == 0) score[w] = acc;
}

// alpha[b,s] = softmax_s(score[s,b])  (one wave per batch row; S=384 -> 12/lane)
__global__ void k_alpha(const float* __restrict__ score, float* __restrict__ alpha) {
  int b = (blockIdx.x << 3) + (threadIdx.x >> 5);
  int lane = threadIdx.x & 31;
  float vals[12];
  float m = -3.4e38f;
  #pragma unroll
  for (int t = 0; t < 12; ++t) {
    vals[t] = score[(lane + 32 * t) * B_N + b];
    m = fmaxf(m, vals[t]);
  }
  m = wave_max(m);
  float sum = 0.f;
  #pragma unroll
  for (int t = 0; t < 12; ++t) { vals[t] = __expf(vals[t] - m); sum += vals[t]; }
  sum = wave_sum(sum);
  float inv = 1.0f / sum;
  #pragma unroll
  for (int t = 0; t < 12; ++t) alpha[b * S_LEN + lane + 32 * t] = vals[t] * inv;
}

// ctx[b,h] = sum_s alpha[b,s] * E[s,b,h]
__global__ void k_ctx(const float* __restrict__ E, const float* __restrict__ alpha,
                      float* __restrict__ ctx) {
  int b = blockIdx.x, h = threadIdx.x;
  float acc = 0.f;
  for (int s = 0; s < S_LEN; ++s)
    acc += alpha[b * S_LEN + s] * E[((size_t)s * B_N + b) * 256 + h];
  ctx[b * 256 + h] = acc;
}

// bf16 weight conversion + initial state copies (h must not alias the input)
__global__ void k_pack(const float* __restrict__ Wih, const float* __restrict__ Whh,
                       const float* __restrict__ Wo, const float* __restrict__ hidden,
                       const float* __restrict__ dec,
                       unsigned short* __restrict__ W1b, unsigned short* __restrict__ Whhb,
                       unsigned short* __restrict__ Wo1b, unsigned short* __restrict__ Wo2b,
                       float* __restrict__ h0, unsigned short* __restrict__ hb0,
                       unsigned short* __restrict__ dbf) {
  int t = blockIdx.x * NTHR + threadIdx.x;          // grid 768*256 = 196608
  {
    int j = t >> 8, k = t & 255;
    W1b[t]  = f2bf(Wih[j * 512 + k]);               // Wih[:, :256]
    Whhb[t] = f2bf(Whh[t]);                         // Whh is [768,256] contiguous
  }
  if (t < 65536) {
    int j = t >> 8, k = t & 255;
    Wo1b[t] = f2bf(Wo[j * 768 + k]);                // Wo[:, :256]
    Wo2b[t] = f2bf(Wo[j * 768 + 256 + k]);          // Wo[:, 256:512]
  }
  if (t < 32768) {
    float hv = hidden[t];
    h0[t] = hv; hb0[t] = f2bf(hv);
    dbf[t] = f2bf(dec[t]);
  }
}

// gctx[b,j] = ctx[b,:] . Wih[j, 256:512] + bih[j]   (98304 entries)
// octx[b,j] = ctx[b,:] . Wo[j, 512:768]  + bo[j]    (32768 entries)
__global__ void k_cctx(const float* __restrict__ ctx, const float* __restrict__ Wih,
                       const float* __restrict__ bih, const float* __restrict__ Wo,
                       const float* __restrict__ bo, float* __restrict__ gctx,
                       float* __restrict__ octx) {
  int w = (blockIdx.x << 3) + (threadIdx.x >> 5);   // 131072 waves
  int lane = threadIdx.x & 31;
  if (w < 98304) {
    int b = w / 768, j = w % 768;
    const float* wr = Wih + j * 512 + 256;
    const float* cb = ctx + b * 256;
    float acc = 0.f;
    for (int k = lane; k < 256; k += 32) acc += cb[k] * wr[k];
    acc = wave_sum(acc);
    if (lane == 0) gctx[w] = acc + bih[j];
  } else {
    int w2 = w - 98304;
    int b = w2 >> 8, j = w2 & 255;
    const float* wr = Wo + j * 768 + 512;
    const float* cb = ctx + b * 256;
    float acc = 0.f;
    for (int k = lane; k < 256; k += 32) acc += cb[k] * wr[k];
    acc = wave_sum(acc);
    if (lane == 0) octx[w2] = acc + bo[j];
  }
}

// ---------------------------------------------------------------------------
// Persistent recurrent kernel: 16 WGs x 256 threads (8 waves).
// Block jt owns output columns [jt*16, jt*16+16) of every GEMM; wave wv owns
// batch rows [wv*16, wv*16+16). LDS holds this block's weight column tiles.
// ---------------------------------------------------------------------------
__global__ __launch_bounds__(NTHR, 1) void decoder_loop(
    const unsigned short* __restrict__ W1b,   // [768,256] bf16 (Wih[:, :256])
    const unsigned short* __restrict__ Whhb,  // [768,256] bf16
    const unsigned short* __restrict__ Wo1b,  // [256,256] bf16
    const unsigned short* __restrict__ Wo2b,  // [256,256] bf16
    const float* __restrict__ gctx,           // [B,768]
    const float* __restrict__ octx,           // [B,256]
    const float* __restrict__ bhh,            // [768]
    float* __restrict__ h0, float* __restrict__ h1,                 // f32 ping-pong
    unsigned short* __restrict__ hb0, unsigned short* __restrict__ hb1, // bf16 ping-pong
    unsigned short* __restrict__ dbf,         // [B,256] bf16
    float* __restrict__ obuf,                 // [B,256]
    float* __restrict__ out,                  // [S,B,V]
    unsigned* bar_cnt, unsigned* bar_gen) {
  // 64KB LDS: 128 rows x 256 bf16. rows 0-47: W1 {r,z,n} slices; 48-95: Whh;
  // 96-111: Wo1; 112-127: Wo2. Each slice row r = global row (base + jt*16 + r).
  __shared__ unsigned short lds[32768];
  const int jt = blockIdx.x;
  const int lane = threadIdx.x & 31;
  const int wv = threadIdx.x >> 5;

  for (int idx = threadIdx.x; idx < 4096; idx += NTHR) {   // 4096 uint4 copies
    int row = idx >> 5, u = (idx & 31) * 8;
    const unsigned short* src;
    if (row < 48)       { int q = row >> 4, r = row & 15;          src = W1b  + (size_t)((q << 8) + (jt << 4) + r) * 256; }
    else if (row < 96)  { int rr = row - 48; int q = rr >> 4, r = rr & 15; src = Whhb + (size_t)((q << 8) + (jt << 4) + r) * 256; }
    else if (row < 112) { int r = row - 96;                        src = Wo1b + (size_t)((jt << 4) + r) * 256; }
    else                { int r = row - 112;                       src = Wo2b + (size_t)((jt << 4) + r) * 256; }
    *(uint4*)&lds[row * 256 + u] = *(const uint4*)&src[u];
  }
  __syncthreads();

  const unsigned short* ldsW1 = lds;             // 3 x [16][256]
  const unsigned short* ldsWh = lds + 48 * 256;  // 3 x [16][256]
  const unsigned short* ldsO1 = lds + 96 * 256;  // [16][256]
  const unsigned short* ldsO2 = lds + 112 * 256; // [16][256]

  const int g = lane >> 4;
  const int j = (jt << 4) + (lane & 15);         // output column (C/D col = lane&15)
  const float bh_r = bhh[j], bh_z = bhh[256 + j], bh_n = bhh[512 + j];

  for (int s = 0; s < S_LEN; ++s) {
    const float*          hR  = (s & 1) ? h1  : h0;
    float*                hW  = (s & 1) ? h0  : h1;
    const unsigned short* hbR = (s & 1) ? hb1 : hb0;
    unsigned short*       hbW = (s & 1) ? hb0 : hb1;
    const int b0 = wv << 4;

    // ---- phase 1: gi/gh GEMMs + GRU gates -> h_new (rows b0.., cols j..)
    {
      v8f aR = {}, aZ = {}, aNi = {}, aNh = {};
      for (int kt = 0; kt < 8; ++kt) {
        int k0 = kt << 5;
        v16bf ad  = load_frag_a(dbf, 256, b0, k0, lane);
        v16bf ah  = load_frag_a(hbR, 256, b0, k0, lane);
        v16bf b1r = load_frag_b(ldsW1,            256, 0, k0, lane);
        v16bf b1z = load_frag_b(ldsW1 + 1 * 4096, 256, 0, k0, lane);
        v16bf b1n = load_frag_b(ldsW1 + 2 * 4096, 256, 0, k0, lane);
        v16bf bhr = load_frag_b(ldsWh,            256, 0, k0, lane);
        v16bf bhz = load_frag_b(ldsWh + 1 * 4096, 256, 0, k0, lane);
        v16bf bhn = load_frag_b(ldsWh + 2 * 4096, 256, 0, k0, lane);
        aR  = wmma_bf16(ad, b1r, aR);  aR = wmma_bf16(ah, bhr, aR);
        aZ  = wmma_bf16(ad, b1z, aZ);  aZ = wmma_bf16(ah, bhz, aZ);
        aNi = wmma_bf16(ad, b1n, aNi);                 // input-gate n part
        aNh = wmma_bf16(ah, bhn, aNh);                 // hidden n part (scaled by r)
      }
      #pragma unroll
      for (int i = 0; i < 8; ++i) {
        int b = b0 + i + (g << 3);                     // C/D row = i + 8*(lane>>4)
        float r  = sigmoidf_(aR[i] + gctx[b * 768 + j] + bh_r);
        float z  = sigmoidf_(aZ[i] + gctx[b * 768 + 256 + j] + bh_z);
        float nn = tanhf_(aNi[i] + gctx[b * 768 + 512 + j] + r * (aNh[i] + bh_n));
        float hn = (1.0f - z) * nn + z * hR[b * 256 + j];
        hW[b * 256 + j]  = hn;
        hbW[b * 256 + j] = f2bf(hn);
      }
    }
    grid_barrier(bar_cnt, bar_gen);

    // ---- phase 2: o = h_new @ Wo1^T + d @ Wo2^T + octx
    {
      v8f aO = {};
      for (int kt = 0; kt < 8; ++kt) {
        int k0 = kt << 5;
        v16bf ahn = load_frag_a(hbW, 256, b0, k0, lane);
        v16bf ad  = load_frag_a(dbf, 256, b0, k0, lane);
        v16bf bo1 = load_frag_b(ldsO1, 256, 0, k0, lane);
        v16bf bo2 = load_frag_b(ldsO2, 256, 0, k0, lane);
        aO = wmma_bf16(ahn, bo1, aO);
        aO = wmma_bf16(ad,  bo2, aO);
      }
      #pragma unroll
      for (int i = 0; i < 8; ++i) {
        int b = b0 + i + (g << 3);
        obuf[b * 256 + j] = aO[i] + octx[b * 256 + j];
      }
    }
    grid_barrier(bar_cnt, bar_gen);

    // ---- phase 3: d_new = softmax(o[b,:]); write out[s] and bf16 state
    {
      int b = (blockIdx.x << 3) + wv;                  // one wave per batch row
      float vv[8];
      #pragma unroll
      for (int t = 0; t < 8; ++t) vv[t] = obuf[b * 256 + lane + 32 * t];
      float mx = vv[0];
      #pragma unroll
      for (int t = 1; t < 8; ++t) mx = fmaxf(mx, vv[t]);
      mx = wave_max(mx);
      float sum = 0.f;
      #pragma unroll
      for (int t = 0; t < 8; ++t) { vv[t] = __expf(vv[t] - mx); sum += vv[t]; }
      sum = wave_sum(sum);
      float inv = 1.0f / sum;
      float* op = out + ((size_t)s * B_N + b) * 256;
      #pragma unroll
      for (int t = 0; t < 8; ++t) {
        float p = vv[t] * inv;
        op[lane + 32 * t]          = p;
        dbf[b * 256 + lane + 32 * t] = f2bf(p);
      }
    }
    grid_barrier(bar_cnt, bar_gen);
  }
}

// ---------------------------------------------------------------------------
extern "C" void kernel_launch(void* const* d_in, const int* in_sizes, int n_in,
                              void* d_out, int out_size, void* d_ws, size_t ws_size,
                              hipStream_t stream) {
  const float* E      = (const float*)d_in[0];   // [S,B,H]
  const float* hidden = (const float*)d_in[1];   // [B,H]
  const float* dec    = (const float*)d_in[2];   // [B,V]
  const float* Wa     = (const float*)d_in[3];   // [H,2H]
  // d_in[4] = ba : unused (softmax shift invariance)
  const float* v      = (const float*)d_in[5];   // [H]
  const float* Wih    = (const float*)d_in[6];   // [3H,2H]
  const float* bih    = (const float*)d_in[7];   // [3H]
  const float* Whh    = (const float*)d_in[8];   // [3H,H]
  const float* bhh    = (const float*)d_in[9];   // [3H]
  const float* Wo     = (const float*)d_in[10];  // [V,3H]
  const float* bo     = (const float*)d_in[11];  // [V]
  float* out = (float*)d_out;

  char* w = (char*)d_ws;
  auto alloc = [&](size_t bytes) { char* p = w; w += (bytes + 255) & ~(size_t)255; return p; };
  unsigned*       bar   = (unsigned*)alloc(256);
  float*          u_e   = (float*)alloc(256 * 4);
  float*          score = (float*)alloc((size_t)S_LEN * B_N * 4);
  float*          alpha = (float*)alloc((size_t)B_N * S_LEN * 4);
  float*          ctx   = (float*)alloc((size_t)B_N * 256 * 4);
  float*          gctx  = (float*)alloc((size_t)B_N * 768 * 4);
  float*          octx  = (float*)alloc((size_t)B_N * 256 * 4);
  unsigned short* W1b   = (unsigned short*)alloc((size_t)768 * 256 * 2);
  unsigned short* Whhb  = (unsigned short*)alloc((size_t)768 * 256 * 2);
  unsigned short* Wo1b  = (unsigned short*)alloc((size_t)256 * 256 * 2);
  unsigned short* Wo2b  = (unsigned short*)alloc((size_t)256 * 256 * 2);
  float*          h0    = (float*)alloc((size_t)B_N * 256 * 4);
  float*          h1    = (float*)alloc((size_t)B_N * 256 * 4);
  unsigned short* hb0   = (unsigned short*)alloc((size_t)B_N * 256 * 2);
  unsigned short* hb1   = (unsigned short*)alloc((size_t)B_N * 256 * 2);
  unsigned short* dbf   = (unsigned short*)alloc((size_t)B_N * 256 * 2);
  float*          obuf  = (float*)alloc((size_t)B_N * 256 * 4);
  (void)in_sizes; (void)n_in; (void)out_size; (void)ws_size;

  hipMemsetAsync(bar, 0, 256, stream);                           // barrier state per launch
  k_ue   <<<1,     NTHR, 0, stream>>>(Wa, v, u_e);
  k_score<<<6144,  NTHR, 0, stream>>>(E, u_e, score);            // 49152 waves
  k_alpha<<<16,    NTHR, 0, stream>>>(score, alpha);             // 128 waves
  k_ctx  <<<128,   NTHR, 0, stream>>>(E, alpha, ctx);
  k_pack <<<768,   NTHR, 0, stream>>>(Wih, Whh, Wo, hidden, dec,
                                      W1b, Whhb, Wo1b, Wo2b, h0, hb0, dbf);
  k_cctx <<<16384, NTHR, 0, stream>>>(ctx, Wih, bih, Wo, bo, gctx, octx);
  decoder_loop<<<NWG, NTHR, 0, stream>>>(W1b, Whhb, Wo1b, Wo2b, gctx, octx, bhh,
                                         h0, h1, hb0, hb1, dbf, obuf, out,
                                         bar, bar + 1);
}
